// STGNN_35673998361303
// MI455X (gfx1250) — matrix operand order repaired and checked by reading, not compiled, not run on hardware
//
#include <hip/hip_runtime.h>
#include <hip/hip_bf16.h>

typedef __attribute__((ext_vector_type(16))) _Float16 v16h;
typedef __attribute__((ext_vector_type(8)))  float    v8f;

#define GAT_H   4
#define GAT_D   128
#define HDIM    128
#define NNODE   100
#define NGRAPH  192     // B*T = 4*48
#define TSTEPS  48
#define BATCH   4
#define MAXE    2048    // >= Etot (1700)

// ===========================================================================
// Operand packing: f32 matrices -> f16 WMMA fragments (ISA 7.12.2 layouts).
// Fragment = 32 lanes x 16 halves (one v16h per lane), 512 halves contiguous.
// Apack index: frag(mt,kc) at ((mt*K32 + kc)*32 + lane)*16
// Bpack index: frag(nt,kc) at ((nt*K32 + kc)*32 + lane)*16
// ===========================================================================

// A [M,K] row-major f32 -> fragments. grid=(K/32, M/16), block=32.
__global__ __launch_bounds__(32)
void pack_a_f16(const float* __restrict__ A, _Float16* __restrict__ Ap, int K) {
    const int kc = blockIdx.x, mt = blockIdx.y, lane = threadIdx.x;
    const int K32 = K >> 5;
    const int l16 = lane & 15, h16 = lane >> 4;
    const int m  = mt * 16 + l16;
    const int kb = kc * 32 + h16 * 8;        // lanes 0-15: K 0..7/16..23 ; 16-31: 8..15/24..31
    const float* arow = A + (size_t)m * K;
    v16h a;
    #pragma unroll
    for (int i = 0; i < 8; ++i) {
        a[i]     = (_Float16)arow[kb + i];
        a[i + 8] = (_Float16)arow[kb + 16 + i];
    }
    *((v16h*)(Ap + (((size_t)mt * K32 + kc) * 32 + lane) * 16)) = a;
}

// B: bTrans==0 -> B is [K,N] row-major; bTrans==1 -> B is [N,K] row-major (use B^T).
// grid=(K/32, N/16), block=32.
__global__ __launch_bounds__(32)
void pack_b_f16(const float* __restrict__ B, _Float16* __restrict__ Bp,
                int K, int N, int bTrans) {
    const int kc = blockIdx.x, nt = blockIdx.y, lane = threadIdx.x;
    const int K32 = K >> 5;
    const int l16 = lane & 15, h16 = lane >> 4;
    const int n  = nt * 16 + l16;
    const int kb = kc * 32 + h16 * 8;        // lanes 0-15: K 0..7/16..23 ; 16-31: 8..15/24..31
    v16h b;
    if (!bTrans) {
        #pragma unroll
        for (int i = 0; i < 16; ++i)
            b[i] = (_Float16)B[(size_t)(kb + i + (i & 8)) * N + n];
    } else {
        const float* brow = B + (size_t)n * K;
        #pragma unroll
        for (int i = 0; i < 16; ++i)
            b[i] = (_Float16)brow[kb + i + (i & 8)];
    }
    *((v16h*)(Bp + (((size_t)nt * K32 + kc) * 32 + lane) * 16)) = b;
}

// ===========================================================================
// C[M,N] = act( A @ B + bias ), operands pre-packed f16 fragments.
// One wave per 32x32 output macro-tile: 2 A-frags x 2 B-frags -> 4 WMMAs per
// K-chunk, pure b128 fragment loads, prefetch of next chunk.
// grid=(N/32, M/32), block=32.  M,N % 32 == 0, K % 32 == 0.
// ===========================================================================
__global__ __launch_bounds__(32)
void gemm_frag_wmma(const _Float16* __restrict__ Ap, const _Float16* __restrict__ Bp,
                    const float* __restrict__ bias, float* __restrict__ C,
                    int M, int N, int K, int doRelu) {
    const int lane = threadIdx.x;
    const int K32  = K >> 5;
    const int mt0  = blockIdx.y * 2;
    const int nt0  = blockIdx.x * 2;

    const v16h* ap0 = (const v16h*)Ap + ((size_t)mt0 * K32) * 32 + lane;
    const v16h* ap1 = ap0 + (size_t)K32 * 32;
    const v16h* bp0 = (const v16h*)Bp + ((size_t)nt0 * K32) * 32 + lane;
    const v16h* bp1 = bp0 + (size_t)K32 * 32;

    v8f acc00 = {}, acc01 = {}, acc10 = {}, acc11 = {};
    for (int kc = 0; kc < K32; ++kc) {
        const int o = kc * 32;
        v16h a0 = ap0[o], a1 = ap1[o];
        v16h b0 = bp0[o], b1 = bp1[o];
        if (kc + 1 < K32) {                       // -> global_prefetch_b8
            __builtin_prefetch((const void*)&ap0[o + 32], 0, 1);
            __builtin_prefetch((const void*)&ap1[o + 32], 0, 1);
            __builtin_prefetch((const void*)&bp0[o + 32], 0, 1);
            __builtin_prefetch((const void*)&bp1[o + 32], 0, 1);
        }
        acc00 = __builtin_amdgcn_wmma_f32_16x16x32_f16(false, a0, false, b0, (short)0, acc00, false, false);
        acc01 = __builtin_amdgcn_wmma_f32_16x16x32_f16(false, a0, false, b1, (short)0, acc01, false, false);
        acc10 = __builtin_amdgcn_wmma_f32_16x16x32_f16(false, a1, false, b0, (short)0, acc10, false, false);
        acc11 = __builtin_amdgcn_wmma_f32_16x16x32_f16(false, a1, false, b1, (short)0, acc11, false, false);
    }

    // C/D layout: VGPR j -> row = tileM*16 + (lane>=16)*8 + j, col = tileN*16 + (lane&15)
    const int l16 = lane & 15, h16 = lane >> 4;
    v8f accs[4] = {acc00, acc01, acc10, acc11};
    #pragma unroll
    for (int s = 0; s < 4; ++s) {
        const int mt = mt0 + (s >> 1);
        const int nt = nt0 + (s & 1);
        const int col = nt * 16 + l16;
        const float bv = bias ? bias[col] : 0.0f;
        #pragma unroll
        for (int j = 0; j < 8; ++j) {
            const int row = mt * 16 + h16 * 8 + j;
            float v = accs[s][j] + bv;
            if (doRelu) v = fmaxf(v, 0.0f);
            C[(size_t)row * N + col] = v;
        }
    }
}

// ---------------------------------------------------------------------------
// Edge attention logits: e[g,e,h] = sum_d leakyrelu(xl[src]+xr[dst]) * att[h,d]
// ---------------------------------------------------------------------------
__global__ void gat_edge_scores(const float* __restrict__ xl, const float* __restrict__ xr,
                                const int* __restrict__ src, const int* __restrict__ dst,
                                const float* __restrict__ att, float* __restrict__ esc,
                                int E, int Etot) {
    int idx = blockIdx.x * blockDim.x + threadIdx.x;
    int total = NGRAPH * Etot * GAT_H;
    if (idx >= total) return;
    int h = idx & (GAT_H - 1);
    int e = (idx >> 2) % Etot;
    int g = idx / (Etot * GAT_H);
    int s = (e < E) ? src[e] : (e - E);   // appended self-loops
    int d = (e < E) ? dst[e] : (e - E);
    const float* pl = xl + ((size_t)(g * NNODE + s) * GAT_H + h) * GAT_D;
    const float* pr = xr + ((size_t)(g * NNODE + d) * GAT_H + h) * GAT_D;
    const float* pa = att + h * GAT_D;
    float acc = 0.0f;
    #pragma unroll 4
    for (int i = 0; i < GAT_D; ++i) {
        float m = pl[i] + pr[i];
        m = (m > 0.0f) ? m : 0.2f * m;    // leaky_relu, slope 0.2
        acc += m * pa[i];
    }
    esc[idx] = acc;
}

// ---------------------------------------------------------------------------
// Per-(graph,node,head): segment max and sum(exp(e - max)) over incoming edges
// ---------------------------------------------------------------------------
__global__ void gat_softmax_stats(const float* __restrict__ esc,
                                  const int* __restrict__ dst,
                                  float* __restrict__ emax, float* __restrict__ denom,
                                  int E, int Etot) {
    int idx = blockIdx.x * blockDim.x + threadIdx.x;
    int total = NGRAPH * NNODE * GAT_H;
    if (idx >= total) return;
    int h = idx & (GAT_H - 1);
    int n = (idx >> 2) % NNODE;
    int g = idx / (NNODE * GAT_H);
    const float* eg = esc + (size_t)g * Etot * GAT_H;
    float mx = -1e30f;
    for (int e = 0; e < Etot; ++e) {
        int d = (e < E) ? dst[e] : (e - E);
        if (d == n) mx = fmaxf(mx, eg[e * GAT_H + h]);
    }
    float s = 0.0f;
    for (int e = 0; e < Etot; ++e) {
        int d = (e < E) ? dst[e] : (e - E);
        if (d == n) s += __expf(eg[e * GAT_H + h] - mx);
    }
    emax[idx] = mx;
    denom[idx] = s;
}

// ---------------------------------------------------------------------------
// out[g,n,d] = relu( mean_h sum_{e:dst=n} alpha[e,h] * xl[g,src_e,h,d] + bias[d] )
// Block = (g,n), 128 threads over d. Edge lists cached in LDS.
// ---------------------------------------------------------------------------
__global__ __launch_bounds__(GAT_D)
void gat_aggregate(const float* __restrict__ xl, const float* __restrict__ esc,
                   const int* __restrict__ src, const int* __restrict__ dst,
                   const float* __restrict__ emax, const float* __restrict__ denom,
                   const float* __restrict__ bias, float* __restrict__ out,
                   int E, int Etot) {
    __shared__ int ssrc[MAXE];
    __shared__ int sdst[MAXE];
    const int g = blockIdx.y;
    const int n = blockIdx.x;
    const int d = threadIdx.x;            // 0..127

    for (int e = d; e < Etot; e += GAT_D) {
        ssrc[e] = (e < E) ? src[e] : (e - E);
        sdst[e] = (e < E) ? dst[e] : (e - E);
    }
    __syncthreads();

    float mx[GAT_H], dn[GAT_H];
    #pragma unroll
    for (int h = 0; h < GAT_H; ++h) {
        size_t si = ((size_t)(g * NNODE + n)) * GAT_H + h;
        mx[h] = emax[si];
        dn[h] = denom[si];
    }
    float acc[GAT_H] = {0.f, 0.f, 0.f, 0.f};
    const float* eg = esc + (size_t)g * Etot * GAT_H;
    for (int e = 0; e < Etot; ++e) {
        if (sdst[e] != n) continue;
        int s = ssrc[e];
        const float* px = xl + (size_t)(g * NNODE + s) * (GAT_H * GAT_D);
        #pragma unroll
        for (int h = 0; h < GAT_H; ++h) {
            float al = __expf(eg[e * GAT_H + h] - mx[h]) / dn[h];
            acc[h] += al * px[h * GAT_D + d];
        }
    }
    float v = 0.25f * (acc[0] + acc[1] + acc[2] + acc[3]) + bias[d];
    v = fmaxf(v, 0.0f);                   // ReLU after each GAT layer
    out[((size_t)(g * NNODE) + n) * GAT_D + d] = v;
}

// ---------------------------------------------------------------------------
// Extract node-0 sequence and transpose [B,T,*] -> [T,B,*]
// ---------------------------------------------------------------------------
__global__ void extract_node0(const float* __restrict__ h2, float* __restrict__ ee) {
    int idx = blockIdx.x * blockDim.x + threadIdx.x;
    if (idx >= TSTEPS * BATCH * HDIM) return;
    int d = idx & (HDIM - 1);
    int b = (idx >> 7) % BATCH;
    int t = idx / (BATCH * HDIM);
    ee[idx] = h2[((size_t)(b * TSTEPS + t) * NNODE + 0) * HDIM + d];
}

// ---------------------------------------------------------------------------
// GRU recurrence. gi = precomputed x@Wi^T+bi, layout [T,B,384].
// Single persistent block, 384 threads; h,gh in LDS; torch gate order (r,z,n).
// ---------------------------------------------------------------------------
__global__ __launch_bounds__(3 * HDIM)
void gru_recurrence(const float* __restrict__ gi, const float* __restrict__ Wh,
                    const float* __restrict__ bh, float* __restrict__ ys,
                    float* __restrict__ hT) {
    __shared__ float hsh[BATCH][HDIM];
    __shared__ float ghsh[BATCH][3 * HDIM];
    const int tid = threadIdx.x;          // 0..383
    for (int i = tid; i < BATCH * HDIM; i += 3 * HDIM) (&hsh[0][0])[i] = 0.0f;
    __syncthreads();

    for (int t = 0; t < TSTEPS; ++t) {
        {
            const float* wrow = Wh + (size_t)tid * HDIM;
            float g0 = bh[tid], g1 = g0, g2 = g0, g3 = g0;
            #pragma unroll 4
            for (int i = 0; i < HDIM; ++i) {
                float w = wrow[i];
                g0 += w * hsh[0][i]; g1 += w * hsh[1][i];
                g2 += w * hsh[2][i]; g3 += w * hsh[3][i];
            }
            ghsh[0][tid] = g0; ghsh[1][tid] = g1;
            ghsh[2][tid] = g2; ghsh[3][tid] = g3;
        }
        __syncthreads();
        if (tid < HDIM) {
            const int i = tid;
            #pragma unroll
            for (int b = 0; b < BATCH; ++b) {
                const float* gib = gi + ((size_t)t * BATCH + b) * (3 * HDIM);
                float r  = 1.0f / (1.0f + __expf(-(gib[i]            + ghsh[b][i])));
                float z  = 1.0f / (1.0f + __expf(-(gib[HDIM + i]     + ghsh[b][HDIM + i])));
                float nn = tanhf(gib[2 * HDIM + i] + r * ghsh[b][2 * HDIM + i]);
                float hnew = (1.0f - z) * nn + z * hsh[b][i];
                hsh[b][i] = hnew;
                if (ys) ys[((size_t)t * BATCH + b) * HDIM + i] = hnew;
            }
        }
        __syncthreads();
    }
    if (tid < HDIM)
        #pragma unroll
        for (int b = 0; b < BATCH; ++b) hT[b * HDIM + tid] = hsh[b][tid];
}

// ---------------------------------------------------------------------------
// Decoder: out[b,q] = (relu(hT@W1 + b1) @ W2 + b2)[b,q]
// ---------------------------------------------------------------------------
__global__ __launch_bounds__(HDIM)
void decoder_mlp(const float* __restrict__ hT, const float* __restrict__ W1,
                 const float* __restrict__ b1, const float* __restrict__ W2,
                 const float* __restrict__ b2, float* __restrict__ out) {
    __shared__ float d1[BATCH][HDIM];
    const int j = threadIdx.x;            // 0..127
    #pragma unroll
    for (int b = 0; b < BATCH; ++b) {
        float acc = b1[j];
        for (int i = 0; i < HDIM; ++i) acc += hT[b * HDIM + i] * W1[i * HDIM + j];
        d1[b][j] = fmaxf(acc, 0.0f);
    }
    __syncthreads();
    if (j < BATCH * 3) {
        int b = j / 3, q = j % 3;
        float acc = b2[q];
        for (int k = 0; k < HDIM; ++k) acc += d1[b][k] * W2[k * 3 + q];
        out[b * 3 + q] = acc;
    }
}

// ---------------------------------------------------------------------------
extern "C" void kernel_launch(void* const* d_in, const int* in_sizes, int n_in,
                              void* d_out, int out_size, void* d_ws, size_t ws_size,
                              hipStream_t stream) {
    (void)n_in; (void)out_size; (void)ws_size;
    const float* x_seq  = (const float*)d_in[0];    // [4,48,100,32]
    const int*   eidx   = (const int*)  d_in[1];    // [2,1600]
    const float* g1_Wl  = (const float*)d_in[2];
    const float* g1_bl  = (const float*)d_in[3];
    const float* g1_Wr  = (const float*)d_in[4];
    const float* g1_br  = (const float*)d_in[5];
    const float* g1_att = (const float*)d_in[6];
    const float* g1_bias= (const float*)d_in[7];
    const float* g2_Wl  = (const float*)d_in[8];
    const float* g2_bl  = (const float*)d_in[9];
    const float* g2_Wr  = (const float*)d_in[10];
    const float* g2_br  = (const float*)d_in[11];
    const float* g2_att = (const float*)d_in[12];
    const float* g2_bias= (const float*)d_in[13];
    const float* gru_Wi0= (const float*)d_in[14];
    const float* gru_Wh0= (const float*)d_in[15];
    const float* gru_bi0= (const float*)d_in[16];
    const float* gru_bh0= (const float*)d_in[17];
    const float* gru_Wi1= (const float*)d_in[18];
    const float* gru_Wh1= (const float*)d_in[19];
    const float* gru_bi1= (const float*)d_in[20];
    const float* gru_bh1= (const float*)d_in[21];
    const float* dec_W1 = (const float*)d_in[22];
    const float* dec_b1 = (const float*)d_in[23];
    const float* dec_W2 = (const float*)d_in[24];
    const float* dec_b2 = (const float*)d_in[25];

    const int E    = in_sizes[1] / 2;      // 1600
    const int Etot = E + NNODE;            // 1700 (with self-loops)
    const int M1   = NGRAPH * NNODE;       // 19200 rows
    const int HW   = GAT_H * GAT_D;        // 512
    const int* src = eidx;
    const int* dst = eidx + E;

    // ---- workspace layout: f32 region, then f16 fragment region ----
    float* ws    = (float*)d_ws;
    float* xl    = ws;  ws += (size_t)M1 * HW;
    float* xr    = ws;  ws += (size_t)M1 * HW;
    float* esc   = ws;  ws += (size_t)NGRAPH * Etot * GAT_H;
    float* emax  = ws;  ws += (size_t)NGRAPH * NNODE * GAT_H;
    float* den   = ws;  ws += (size_t)NGRAPH * NNODE * GAT_H;
    float* h1    = ws;  ws += (size_t)M1 * HDIM;
    float* h2    = ws;  ws += (size_t)M1 * HDIM;
    float* ee    = ws;  ws += (size_t)NGRAPH * HDIM;
    float* gi0   = ws;  ws += (size_t)NGRAPH * 3 * HDIM;
    float* y0    = ws;  ws += (size_t)NGRAPH * HDIM;
    float* gi1   = ws;  ws += (size_t)NGRAPH * 3 * HDIM;
    float* hT0   = ws;  ws += (size_t)BATCH * HDIM;
    float* hT1   = ws;  ws += (size_t)BATCH * HDIM;

    _Float16* hp = (_Float16*)ws;          // 32B-aligned (all f32 counts % 8 == 0)
    _Float16* apX  = hp;  hp += (size_t)M1 * 32;          // x packed     (K=32)
    _Float16* apH  = hp;  hp += (size_t)M1 * HDIM;        // h1 packed    (K=128)
    _Float16* apE  = hp;  hp += (size_t)NGRAPH * HDIM;    // ee packed
    _Float16* apY  = hp;  hp += (size_t)NGRAPH * HDIM;    // y0 packed
    _Float16* bp1l = hp;  hp += (size_t)32  * HW;
    _Float16* bp1r = hp;  hp += (size_t)32  * HW;
    _Float16* bp2l = hp;  hp += (size_t)HDIM * HW;
    _Float16* bp2r = hp;  hp += (size_t)HDIM * HW;
    _Float16* bpW0 = hp;  hp += (size_t)HDIM * 3 * HDIM;
    _Float16* bpW1 = hp;  hp += (size_t)HDIM * 3 * HDIM;

    const dim3 wave(32);
    const int scoreThreads = NGRAPH * Etot * GAT_H;
    const int statThreads  = NGRAPH * NNODE * GAT_H;

    // ---- pack all weights once ----
    pack_b_f16<<<dim3(1, HW/16), wave, 0, stream>>>(g1_Wl, bp1l, 32, HW, 0);
    pack_b_f16<<<dim3(1, HW/16), wave, 0, stream>>>(g1_Wr, bp1r, 32, HW, 0);
    pack_b_f16<<<dim3(HDIM/32, HW/16), wave, 0, stream>>>(g2_Wl, bp2l, HDIM, HW, 0);
    pack_b_f16<<<dim3(HDIM/32, HW/16), wave, 0, stream>>>(g2_Wr, bp2r, HDIM, HW, 0);
    pack_b_f16<<<dim3(HDIM/32, (3*HDIM)/16), wave, 0, stream>>>(gru_Wi0, bpW0, HDIM, 3*HDIM, 1);
    pack_b_f16<<<dim3(HDIM/32, (3*HDIM)/16), wave, 0, stream>>>(gru_Wi1, bpW1, HDIM, 3*HDIM, 1);

    // ---- GAT layer 1 (Fin = 32): pack x once, two WMMA GEMMs share it ----
    pack_a_f16<<<dim3(1, M1/16), wave, 0, stream>>>(x_seq, apX, 32);
    gemm_frag_wmma<<<dim3(HW/32, M1/32), wave, 0, stream>>>(apX, bp1l, g1_bl, xl, M1, HW, 32, 0);
    gemm_frag_wmma<<<dim3(HW/32, M1/32), wave, 0, stream>>>(apX, bp1r, g1_br, xr, M1, HW, 32, 0);
    gat_edge_scores<<<(scoreThreads + 255)/256, 256, 0, stream>>>(xl, xr, src, dst, g1_att, esc, E, Etot);
    gat_softmax_stats<<<(statThreads + 255)/256, 256, 0, stream>>>(esc, dst, emax, den, E, Etot);
    gat_aggregate<<<dim3(NNODE, NGRAPH), GAT_D, 0, stream>>>(xl, esc, src, dst, emax, den, g1_bias, h1, E, Etot);

    // ---- GAT layer 2 (Fin = 128) ----
    pack_a_f16<<<dim3(HDIM/32, M1/16), wave, 0, stream>>>(h1, apH, HDIM);
    gemm_frag_wmma<<<dim3(HW/32, M1/32), wave, 0, stream>>>(apH, bp2l, g2_bl, xl, M1, HW, HDIM, 0);
    gemm_frag_wmma<<<dim3(HW/32, M1/32), wave, 0, stream>>>(apH, bp2r, g2_br, xr, M1, HW, HDIM, 0);
    gat_edge_scores<<<(scoreThreads + 255)/256, 256, 0, stream>>>(xl, xr, src, dst, g2_att, esc, E, Etot);
    gat_softmax_stats<<<(statThreads + 255)/256, 256, 0, stream>>>(esc, dst, emax, den, E, Etot);
    gat_aggregate<<<dim3(NNODE, NGRAPH), GAT_D, 0, stream>>>(xl, esc, src, dst, emax, den, g2_bias, h2, E, Etot);

    // ---- node-0 sequence, [T,B,128] ----
    extract_node0<<<(TSTEPS*BATCH*HDIM + 255)/256, 256, 0, stream>>>(h2, ee);

    // ---- GRU layer 0: gi = ee @ Wi0^T + bi0 (WMMA), then recurrence ----
    pack_a_f16<<<dim3(HDIM/32, NGRAPH/16), wave, 0, stream>>>(ee, apE, HDIM);
    gemm_frag_wmma<<<dim3((3*HDIM)/32, NGRAPH/32), wave, 0, stream>>>(apE, bpW0, gru_bi0, gi0, NGRAPH, 3*HDIM, HDIM, 0);
    gru_recurrence<<<1, 3*HDIM, 0, stream>>>(gi0, gru_Wh0, gru_bh0, y0, hT0);

    // ---- GRU layer 1 ----
    pack_a_f16<<<dim3(HDIM/32, NGRAPH/16), wave, 0, stream>>>(y0, apY, HDIM);
    gemm_frag_wmma<<<dim3((3*HDIM)/32, NGRAPH/32), wave, 0, stream>>>(apY, bpW1, gru_bi1, gi1, NGRAPH, 3*HDIM, HDIM, 0);
    gru_recurrence<<<1, 3*HDIM, 0, stream>>>(gi1, gru_Wh1, gru_bh1, nullptr, hT1);

    // ---- decoder ----
    decoder_mlp<<<1, HDIM, 0, stream>>>(hT1, dec_W1, dec_b1, dec_W2, dec_b2, (float*)d_out);
}
